// FGALlamaAttention_81320910782493
// MI455X (gfx1250) — matrix operand order repaired and checked by reading, not compiled, not run on hardware
//
#include <hip/hip_runtime.h>

// ---------------------------------------------------------------------------
// Problem constants (match the JAX reference)
// ---------------------------------------------------------------------------
constexpr int BATCH = 2;
constexpr int SEQ   = 2048;
constexpr int HDIM  = 3072;   // hidden size
constexpr int NH    = 24;     // query heads
constexpr int NKV   = 8;      // kv heads
constexpr int HD    = 128;    // head dim
constexpr int QOUT  = NH * HD;   // 3072
constexpr int KVOUT = NKV * HD;  // 1024

typedef __attribute__((ext_vector_type(16))) __bf16 v16bf;
typedef __attribute__((ext_vector_type(8)))  float  v8f;

union FragU { v16bf v; unsigned int u[8]; };

__device__ __forceinline__ unsigned short f32_to_bf16(float f) {
  unsigned int u = __float_as_uint(f);
  u += 0x7FFFu + ((u >> 16) & 1u);   // round-to-nearest-even
  return (unsigned short)(u >> 16);
}

// A-matrix 16x32 bf16 fragment (wave32): lane owns row M = lane%16; element
// pair j (one VGPR) covers k = (j<4 ? 2j : 2j+8) + 8*(lane>=16).
__device__ __forceinline__ v16bf load_frag_a(const unsigned short* p, int hi8) {
  FragU f;
#pragma unroll
  for (int j = 0; j < 8; ++j) {
    int k = ((j < 4) ? (2 * j) : (2 * j + 8)) + hi8;
    f.u[j] = *reinterpret_cast<const unsigned int*>(p + k);
  }
  return f.v;
}

// B-matrix 32x16 bf16 fragment: lane owns col N = lane%16; element pair j
// covers k = (j<4 ? 2j : 2j+8) + 8*(lane>=16) (same order as A).
__device__ __forceinline__ v16bf load_frag_b(const unsigned short* p, int hi16) {
  FragU f;
#pragma unroll
  for (int j = 0; j < 8; ++j)
    f.u[j] = *reinterpret_cast<const unsigned int*>(p + ((j < 4) ? (2 * j) : (2 * j + 8)) + (hi16 >> 1));
  return f.v;
}

__device__ __forceinline__ float fast_exp2(float x) {
#if __has_builtin(__builtin_amdgcn_exp2f)
  return __builtin_amdgcn_exp2f(x);
#else
  return exp2f(x);
#endif
}

// ---------------------------------------------------------------------------
// f32 -> bf16 cast (grid-stride)
// ---------------------------------------------------------------------------
__global__ void cast_f32_bf16(const float* __restrict__ in,
                              unsigned short* __restrict__ out, long long n) {
  long long i = (long long)blockIdx.x * blockDim.x + threadIdx.x;
  long long stride = (long long)gridDim.x * blockDim.x;
  for (; i < n; i += stride) out[i] = f32_to_bf16(in[i]);
}

// ---------------------------------------------------------------------------
// bf16 WMMA GEMM: C[M,N] = A[M,K] * W[K,N]   (row-major, dims % tile == 0)
// Block tile 128x128, 8 waves; register-prefetch software pipeline over K.
// ---------------------------------------------------------------------------
constexpr int GTM = 128, GTN = 128, GTK = 32;
constexpr int GPITCH = 40;  // halfwords; 80B rows keep 16B-aligned b128 stores

template <bool STORE_F32>
__global__ __launch_bounds__(256)
void gemm_bf16_wmma(const unsigned short* __restrict__ A,
                    const unsigned short* __restrict__ W,
                    float* __restrict__ Cf, unsigned short* __restrict__ Cb,
                    int M, int N, int K) {
  __shared__ unsigned short As[GTM * GPITCH];       // [row][k]
  __shared__ unsigned short Wt[GTN * GPITCH];       // [n][k]  (transposed)

  const int tid  = threadIdx.x;
  const int lane = tid & 31;
  const int wave = tid >> 5;
  const int ln16 = lane & 15;
  const int hi8  = (lane >> 4) << 3;
  const int hi16 = (lane >> 4) << 4;
  const int bm   = blockIdx.y * GTM;
  const int bn   = blockIdx.x * GTN;

  // per-thread staging coordinates
  const int arow = tid >> 1, aseg = (tid & 1) << 4;   // A: 128 rows x 2 segs
  const int wk   = tid >> 3, wn0  = (tid & 7) << 4;   // W: 32 k  x 8 segs
  const unsigned short* ag = A + (long long)(bm + arow) * K + aseg;
  const unsigned short* wg = W + (long long)wk * N + bn + wn0;

  v8f acc[8];
#pragma unroll
  for (int t = 0; t < 8; ++t)
#pragma unroll
    for (int r = 0; r < 8; ++r) acc[t][r] = 0.0f;

  // prologue: fetch K-step 0 into registers
  uint4 a0 = *reinterpret_cast<const uint4*>(ag);
  uint4 a1 = *reinterpret_cast<const uint4*>(ag + 8);
  uint4 w0 = *reinterpret_cast<const uint4*>(wg);
  uint4 w1 = *reinterpret_cast<const uint4*>(wg + 8);

  for (int k0 = 0; k0 < K; k0 += GTK) {
    __syncthreads();
    *reinterpret_cast<uint4*>(&As[arow * GPITCH + aseg])     = a0;
    *reinterpret_cast<uint4*>(&As[arow * GPITCH + aseg + 8]) = a1;
    {
      union { uint4 q[2]; unsigned short s[16]; } v;
      v.q[0] = w0; v.q[1] = w1;
#pragma unroll
      for (int i = 0; i < 16; ++i) Wt[(wn0 + i) * GPITCH + wk] = v.s[i];
    }
    __syncthreads();

    // issue next K-step's global loads; they complete during the WMMAs below
    if (k0 + GTK < K) {
      ag += GTK;
      wg += (long long)GTK * N;
      a0 = *reinterpret_cast<const uint4*>(ag);
      a1 = *reinterpret_cast<const uint4*>(ag + 8);
      w0 = *reinterpret_cast<const uint4*>(wg);
      w1 = *reinterpret_cast<const uint4*>(wg + 8);
    }

    // batch all fragment loads, then a dense WMMA chain
    v16bf af = load_frag_a(&As[(wave * 16 + ln16) * GPITCH], hi8);
    v16bf bf[8];
#pragma unroll
    for (int t = 0; t < 8; ++t)
      bf[t] = load_frag_b(&Wt[(t * 16 + ln16) * GPITCH], hi16);
#pragma unroll
    for (int t = 0; t < 8; ++t)
      acc[t] = __builtin_amdgcn_wmma_f32_16x16x32_bf16(
          false, af, false, bf[t], (short)0, acc[t], false, false);
  }

  // Epilogue: lane holds (M = r + hi8, N = t*16 + ln16); 32-bit index math
  const int row0 = bm + wave * 16 + hi8;
  const int col0 = bn + ln16;
#pragma unroll
  for (int r = 0; r < 8; ++r) {
    const int base = (row0 + r) * N + col0;
#pragma unroll
    for (int t = 0; t < 8; ++t) {
      if (STORE_F32) Cf[base + t * 16] = acc[t][r];
      else           Cb[base + t * 16] = f32_to_bf16(acc[t][r]);
    }
  }
}

// ---------------------------------------------------------------------------
// Flash attention (GQA). grid = (SEQ/128, NH, BATCH), 256 threads.
// Wave w owns Q rows [q0, q0+16); all 8 waves share one KV head's K/V chunks
// staged in LDS (register-prefetched). Online softmax in base-2.
// ---------------------------------------------------------------------------
constexpr int KC = 32;        // keys per chunk
constexpr int KP = 136;       // Kl pitch [key][d]  (272B rows: b128-aligned)
constexpr int VP = 40;        // Vl pitch [d][key]
constexpr int PP = 40;        // Pl pitch [q][key]

__global__ __launch_bounds__(256)
void attn_fwd(const unsigned short* __restrict__ Q,
              const unsigned short* __restrict__ Km,
              const unsigned short* __restrict__ Vm,
              const float* __restrict__ mask,
              unsigned short* __restrict__ ctx) {
  __shared__ unsigned short Kl[KC * KP];
  __shared__ unsigned short Vl[HD * VP];
  __shared__ unsigned short Pl[8 * 16 * PP];

  const int tid  = threadIdx.x;
  const int lane = tid & 31;
  const int wave = tid >> 5;
  const int ln16 = lane & 15;
  const int hi8  = (lane >> 4) << 3;
  const int hi16 = (lane >> 4) << 4;
  const int b = blockIdx.z, h = blockIdx.y;
  const int q0 = blockIdx.x * 128 + wave * 16;
  const int kvh = h / (NH / NKV);

  const unsigned short* Qb = Q  + ((long long)b * SEQ) * QOUT  + h   * HD;
  const unsigned short* Kb = Km + ((long long)b * SEQ) * KVOUT + kvh * HD;
  const unsigned short* Vb = Vm + ((long long)b * SEQ) * KVOUT + kvh * HD;
  const float* Mb = mask + (long long)b * SEQ * SEQ;
  unsigned short* Ob = ctx + ((long long)b * SEQ) * QOUT + h * HD;

  // Q fragments (16 rows x 128 d) live in registers for the whole loop.
  v16bf qf[4];
#pragma unroll
  for (int dk = 0; dk < 4; ++dk)
    qf[dk] = load_frag_a(Qb + (long long)(q0 + ln16) * QOUT + dk * 32, hi8);

  float m[8], l[8];
  v8f o[8];
#pragma unroll
  for (int r = 0; r < 8; ++r) { m[r] = -1e30f; l[r] = 0.0f; }
#pragma unroll
  for (int t = 0; t < 8; ++t)
#pragma unroll
    for (int r = 0; r < 8; ++r) o[t][r] = 0.0f;

  const float L2E = 1.4426950408889634f;
  const float SC  = 0.08838834764831845f * L2E;   // 1/sqrt(128) * log2(e)

  // per-thread KV staging coordinates + chunk-0 register prefetch
  const int srow = tid >> 3;          // 0..31 (key)
  const int sseg = (tid & 7) << 4;    // 0..112 (d)
  const unsigned short* gk = Kb + srow * KVOUT + sseg;
  const unsigned short* gv = Vb + srow * KVOUT + sseg;
  uint4 kr0 = *reinterpret_cast<const uint4*>(gk);
  uint4 kr1 = *reinterpret_cast<const uint4*>(gk + 8);
  uint4 vr0 = *reinterpret_cast<const uint4*>(gv);
  uint4 vr1 = *reinterpret_cast<const uint4*>(gv + 8);

  for (int kc = 0; kc < SEQ; kc += KC) {
    __syncthreads();
    // stage K chunk [32][128] row-major and V chunk transposed [128][32]
    *reinterpret_cast<uint4*>(&Kl[srow * KP + sseg])     = kr0;
    *reinterpret_cast<uint4*>(&Kl[srow * KP + sseg + 8]) = kr1;
    {
      union { uint4 q[2]; unsigned short s[16]; } vv;
      vv.q[0] = vr0; vv.q[1] = vr1;
#pragma unroll
      for (int i = 0; i < 16; ++i) Vl[(sseg + i) * VP + srow] = vv.s[i];
    }
    __syncthreads();

    // issue next chunk's global loads; they complete during score math below
    if (kc + KC < SEQ) {
      gk += KC * KVOUT;
      gv += KC * KVOUT;
      kr0 = *reinterpret_cast<const uint4*>(gk);
      kr1 = *reinterpret_cast<const uint4*>(gk + 8);
      vr0 = *reinterpret_cast<const uint4*>(gv);
      vr1 = *reinterpret_cast<const uint4*>(gv + 8);
    }

    // hoist mask loads so they overlap the WMMAs
    float mv[2][8];
#pragma unroll
    for (int t = 0; t < 2; ++t)
#pragma unroll
      for (int r = 0; r < 8; ++r)
        mv[t][r] = Mb[(q0 + r + hi8) * SEQ + kc + t * 16 + ln16];

    // scores: S[16q x 32k] via two 16-key tiles, contraction over d = 128
    v16bf kf[8];
#pragma unroll
    for (int t = 0; t < 2; ++t)
#pragma unroll
      for (int dk = 0; dk < 4; ++dk)
        kf[t * 4 + dk] =
            load_frag_b(&Kl[(t * 16 + ln16) * KP + dk * 32], hi16);

    v8f sc[2];
#pragma unroll
    for (int t = 0; t < 2; ++t) {
      v8f a;
#pragma unroll
      for (int r = 0; r < 8; ++r) a[r] = 0.0f;
#pragma unroll
      for (int dk = 0; dk < 4; ++dk)
        a = __builtin_amdgcn_wmma_f32_16x16x32_bf16(
            false, qf[dk], false, kf[t * 4 + dk], (short)0, a, false, false);
      sc[t] = a;
    }
#pragma unroll
    for (int t = 0; t < 2; ++t)
#pragma unroll
      for (int r = 0; r < 8; ++r)
        sc[t][r] = sc[t][r] * SC + mv[t][r] * L2E;   // base-2 domain

    // online softmax: row (r + hi8) lives in one 16-lane half-wave
    float alpha[8];
#pragma unroll
    for (int r = 0; r < 8; ++r) {
      float cm = fmaxf(sc[0][r], sc[1][r]);
#pragma unroll
      for (int x = 1; x < 16; x <<= 1) cm = fmaxf(cm, __shfl_xor(cm, x, 32));
      float mn = fmaxf(m[r], cm);
      alpha[r] = fast_exp2(m[r] - mn);
      m[r] = mn;
      float p0 = fast_exp2(sc[0][r] - mn);
      float p1 = fast_exp2(sc[1][r] - mn);
      float ls = p0 + p1;
#pragma unroll
      for (int x = 1; x < 16; x <<= 1) ls += __shfl_xor(ls, x, 32);
      l[r] = l[r] * alpha[r] + ls;
      // stash probabilities (C-layout -> A-layout via LDS)
      unsigned short* pr = &Pl[(wave * 16 + r + hi8) * PP];
      pr[ln16]      = f32_to_bf16(p0);
      pr[16 + ln16] = f32_to_bf16(p1);
    }
#pragma unroll
    for (int t = 0; t < 8; ++t)
#pragma unroll
      for (int r = 0; r < 8; ++r) o[t][r] *= alpha[r];
    __syncthreads();

    // O += P[16x32] @ V[32x128]: batch fragment loads, dense WMMA chain
    v16bf pf = load_frag_a(&Pl[(wave * 16 + ln16) * PP], hi8);
    v16bf vf[8];
#pragma unroll
    for (int t = 0; t < 8; ++t)
      vf[t] = load_frag_b(&Vl[(t * 16 + ln16) * VP], hi16);
#pragma unroll
    for (int t = 0; t < 8; ++t)
      o[t] = __builtin_amdgcn_wmma_f32_16x16x32_bf16(
          false, pf, false, vf[t], (short)0, o[t], false, false);
  }

  // normalize + store context (bf16) in [B,S,NH*HD] layout
#pragma unroll
  for (int r = 0; r < 8; ++r) l[r] = 1.0f / l[r];
#pragma unroll
  for (int r = 0; r < 8; ++r) {
    const int base = (q0 + r + hi8) * QOUT + ln16;
#pragma unroll
    for (int t = 0; t < 8; ++t)
      Ob[base + t * 16] = f32_to_bf16(o[t][r] * l[r]);
  }
}

// ---------------------------------------------------------------------------
// Host-side launch
// ---------------------------------------------------------------------------
extern "C" void kernel_launch(void* const* d_in, const int* in_sizes, int n_in,
                              void* d_out, int out_size, void* d_ws, size_t ws_size,
                              hipStream_t stream) {
  const float* hidden = (const float*)d_in[0];
  const float* amask  = (const float*)d_in[1];
  const float* Wq     = (const float*)d_in[2];
  const float* Wk     = (const float*)d_in[3];
  const float* Wv     = (const float*)d_in[4];
  const float* Wo     = (const float*)d_in[5];
  float* out = (float*)d_out;

  const long long M = (long long)BATCH * SEQ;   // 4096 token rows

  // carve bf16 scratch out of d_ws (256B aligned slices)
  char* ws = (char*)d_ws;
  auto carve = [&](long long elems) {
    unsigned short* p = (unsigned short*)ws;
    ws += ((elems * 2 + 255) / 256) * 256;
    return p;
  };
  unsigned short* hid_b = carve(M * HDIM);
  unsigned short* wq_b  = carve((long long)HDIM * QOUT);
  unsigned short* wk_b  = carve((long long)HDIM * KVOUT);
  unsigned short* wv_b  = carve((long long)HDIM * KVOUT);
  unsigned short* wo_b  = carve((long long)QOUT * HDIM);
  unsigned short* q_b   = carve(M * QOUT);
  unsigned short* k_b   = carve(M * KVOUT);
  unsigned short* v_b   = carve(M * KVOUT);
  unsigned short* ctx_b = carve(M * QOUT);

  // 1) casts
  cast_f32_bf16<<<2048, 256, 0, stream>>>(hidden, hid_b, M * HDIM);
  cast_f32_bf16<<<2048, 256, 0, stream>>>(Wq, wq_b, (long long)HDIM * QOUT);
  cast_f32_bf16<<<1024, 256, 0, stream>>>(Wk, wk_b, (long long)HDIM * KVOUT);
  cast_f32_bf16<<<1024, 256, 0, stream>>>(Wv, wv_b, (long long)HDIM * KVOUT);
  cast_f32_bf16<<<2048, 256, 0, stream>>>(Wo, wo_b, (long long)QOUT * HDIM);

  // 2) projections (bf16 out)
  gemm_bf16_wmma<false><<<dim3(QOUT / GTN,  M / GTM), 256, 0, stream>>>(
      hid_b, wq_b, nullptr, q_b, (int)M, QOUT, HDIM);
  gemm_bf16_wmma<false><<<dim3(KVOUT / GTN, M / GTM), 256, 0, stream>>>(
      hid_b, wk_b, nullptr, k_b, (int)M, KVOUT, HDIM);
  gemm_bf16_wmma<false><<<dim3(KVOUT / GTN, M / GTM), 256, 0, stream>>>(
      hid_b, wv_b, nullptr, v_b, (int)M, KVOUT, HDIM);

  // 3) attention core
  attn_fwd<<<dim3(SEQ / 128, NH, BATCH), 256, 0, stream>>>(
      q_b, k_b, v_b, amask, ctx_b);

  // 4) output projection (f32 out -> d_out)
  gemm_bf16_wmma<true><<<dim3(HDIM / GTN, M / GTM), 256, 0, stream>>>(
      ctx_b, wo_b, out, nullptr, (int)M, HDIM, QOUT);

  (void)in_sizes; (void)n_in; (void)out_size; (void)ws_size;
}
